// CSPNet_71657234367119
// MI455X (gfx1250) — matrix operand order repaired and checked by reading, not compiled, not run on hardware
//
#include <hip/hip_runtime.h>
#include <hip/hip_bf16.h>

// Problem sizes (fixed by the reference)
#define NATOMS  8192
#define NGRAPH  256
#define NEDGE   163840
#define HDIM    128
#define ZDIM    256
#define NLAYER  4
#define EDGE_IN 325
#define EDGE_KP 352   // 11 * 32 (bf16 WMMA K-tile)
#define NODE_KP 256   // 8 * 32
#define LAT_KP  384   // 12 * 32

typedef __attribute__((ext_vector_type(16))) __bf16          v16bf;
typedef __attribute__((ext_vector_type(8)))  float           v8f;
typedef __attribute__((ext_vector_type(8)))  unsigned short  v8u16;
typedef __attribute__((ext_vector_type(16))) unsigned short  v16u16;
typedef __attribute__((ext_vector_type(4)))  unsigned short  v4u16;
typedef __attribute__((ext_vector_type(2)))  unsigned short  v2u16;

__device__ __forceinline__ unsigned short f2bf(float f) {
  unsigned int u = __float_as_uint(f);
  u += 0x7fffu + ((u >> 16) & 1u);          // round-to-nearest-even
  return (unsigned short)(u >> 16);
}
__device__ __forceinline__ float silu_f(float x) { return x / (1.0f + __expf(-x)); }

// ---------------------------------------------------------------------------
// Wave-level GEMM: 16 rows (staged row-major bf16 in LDS) x (K, 128) packed
// weights -> 8 accumulated 16x16 f32 fragments.
// A layout (16x32 bf16): lane_lo = M; elements 0..15 -> K =
// kt*32 + 16*hs + {0..15}.  B layout (32x16 bf16): lane_lo
// = N; elements 0..15 -> K = kt*32 + 16*hs + {0..15} (pre-packed that way).
// ---------------------------------------------------------------------------
template <int KT>
__device__ __forceinline__ void wave_gemm8(const unsigned short* st, int strideH,
                                           const unsigned short* __restrict__ wp,
                                           v8f acc[8], int lane) {
  const int lo = lane & 15;
  const int hs = lane >> 4;
  for (int kt = 0; kt < KT; ++kt) {
    const unsigned short* ap = st + lo * strideH + kt * 32 + hs * 16;
    v8u16 a0 = *(const v8u16*)(ap);
    v8u16 a1 = *(const v8u16*)(ap + 8);
    v16u16 acomb = __builtin_shufflevector(a0, a1, 0,1,2,3,4,5,6,7,8,9,10,11,12,13,14,15);
    v16bf av = __builtin_bit_cast(v16bf, acomb);
#pragma unroll
    for (int nt = 0; nt < 8; ++nt) {
      const unsigned short* bp = wp + (size_t)(((kt << 3) + nt) * 32 + lane) * 16;
      v8u16 b0 = *(const v8u16*)(bp);
      v8u16 b1 = *(const v8u16*)(bp + 8);
      v16u16 bcomb = __builtin_shufflevector(b0, b1, 0,1,2,3,4,5,6,7,8,9,10,11,12,13,14,15);
      v16bf bv = __builtin_bit_cast(v16bf, bcomb);
      acc[nt] = __builtin_amdgcn_wmma_f32_16x16x32_bf16(
          false, av, false, bv, (short)0, acc[nt], false, false);
    }
  }
}

// ---------------------------------------------------------------------------
// Weight packing: W (Kact x 128) fp32 row-major -> bf16 B-fragment stream.
// Element index within a matrix: [kt][nt][lane][e] ; k = kt*32 + 16*hs + e.
// grid.y = matrix index (layer).
// ---------------------------------------------------------------------------
__global__ void pack_kernel(const float* __restrict__ W, unsigned short* __restrict__ out,
                            int Kact, int ktiles) {
  const int mat = blockIdx.y;
  const int per = ktiles * 4096;
  int idx = blockIdx.x * blockDim.x + threadIdx.x;
  if (idx >= per) return;
  const int e    = idx & 15;
  const int lane = (idx >> 4) & 31;
  const int nt   = (idx >> 9) & 7;
  const int kt   = idx >> 12;
  const int k = kt * 32 + (lane >> 4) * 16 + e;
  const int n = nt * 16 + (lane & 15);
  float v = (k < Kact) ? W[(size_t)mat * Kact * HDIM + (size_t)k * HDIM + n] : 0.0f;
  out[(size_t)mat * per + idx] = f2bf(v);
}

// --------------------------- small prep kernels ----------------------------
__global__ void dis_kernel(const float* __restrict__ frac, const int* __restrict__ src,
                           const int* __restrict__ dst, float* __restrict__ dis) {
  int e = blockIdx.x * blockDim.x + threadIdx.x;
  if (e >= NEDGE) return;
  int s = src[e], d = dst[e];
#pragma unroll
  for (int c = 0; c < 3; ++c) {
    float df = frac[d * 3 + c] - frac[s * 3 + c];
    df -= floorf(df);                         // python % 1.0
#pragma unroll
    for (int f = 0; f < 10; ++f) {
      float ang = df * (6.28318530717958647692f * (float)f);
      float sv, cv;
      __sincosf(ang, &sv, &cv);
      dis[(size_t)e * 60 + c * 10 + f]      = sv;
      dis[(size_t)e * 60 + 30 + c * 10 + f] = cv;
    }
  }
}

__global__ void latips_kernel(const float* __restrict__ lat, float* __restrict__ out) {
  int idx = blockIdx.x * blockDim.x + threadIdx.x;
  if (idx >= NGRAPH * 9) return;
  int b = idx / 9, r = idx % 9, i = r / 3, k = r % 3;
  float s = 0.f;
#pragma unroll
  for (int j = 0; j < 3; ++j) s += lat[b * 9 + i * 3 + j] * lat[b * 9 + k * 3 + j];
  out[idx] = s;                               // einsum('bij,bkj->bik') flattened
}

// Adapter MLP (tiny): c[l,b,:] = silu(silu(cemb*w1+b1)@w2+b2)@mixin
__global__ __launch_bounds__(HDIM) void adapt_kernel(
    const float* __restrict__ cemb, const float* __restrict__ aw1,
    const float* __restrict__ ab1, const float* __restrict__ aw2,
    const float* __restrict__ ab2, const float* __restrict__ mixin,
    float* __restrict__ cbuf) {
  const int l = blockIdx.x / NGRAPH;
  const int b = blockIdx.x % NGRAPH;
  const int j = threadIdx.x;
  __shared__ float t[HDIM];
  __shared__ float u[HDIM];
  float ce = cemb[b];
  t[j] = silu_f(ce * aw1[l * HDIM + j] + ab1[l * HDIM + j]);
  __syncthreads();
  float s = ab2[l * HDIM + j];
  for (int c = 0; c < HDIM; ++c) s += t[c] * aw2[((size_t)l * HDIM + c) * HDIM + j];
  u[j] = silu_f(s);
  __syncthreads();
  float o = 0.f;
  for (int c = 0; c < HDIM; ++c) o += u[c] * mixin[((size_t)l * HDIM + c) * HDIM + j];
  cbuf[((size_t)l * NGRAPH + b) * HDIM + j] = o;
}

// --------------------------- h0 = concat(emb, z) @ W_latent + b ------------
__global__ __launch_bounds__(128) void h0_kernel(
    const int* __restrict__ atom_types, const float* __restrict__ emb,
    const float* __restrict__ z, const int* __restrict__ n2g,
    const unsigned short* __restrict__ pwlat, const float* __restrict__ blat,
    float* __restrict__ h) {
  __shared__ unsigned short stage[4][16 * LAT_KP];
  const int wave = threadIdx.x >> 5;
  const int lane = threadIdx.x & 31;
  const int lo = lane & 15, hs = lane >> 4;
  const int n0 = (blockIdx.x * 4 + wave) * 16;
  unsigned short* st = stage[wave];

  for (int e = 0; e < 16; ++e) {
    const int node = n0 + e;
    const int at = atom_types[node];
    const int g = n2g[node];
    float4 ev = *(const float4*)(emb + (size_t)at * HDIM + lane * 4);
    v4u16 ep = { f2bf(ev.x), f2bf(ev.y), f2bf(ev.z), f2bf(ev.w) };
    *(v4u16*)(st + e * LAT_KP + lane * 4) = ep;
    float4 z0 = *(const float4*)(z + (size_t)g * ZDIM + lane * 8);
    float4 z1 = *(const float4*)(z + (size_t)g * ZDIM + lane * 8 + 4);
    v4u16 zp0 = { f2bf(z0.x), f2bf(z0.y), f2bf(z0.z), f2bf(z0.w) };
    v4u16 zp1 = { f2bf(z1.x), f2bf(z1.y), f2bf(z1.z), f2bf(z1.w) };
    *(v4u16*)(st + e * LAT_KP + HDIM + lane * 8) = zp0;
    *(v4u16*)(st + e * LAT_KP + HDIM + lane * 8 + 4) = zp1;
  }
  __syncthreads();

  const v8f vz = {0.f,0.f,0.f,0.f,0.f,0.f,0.f,0.f};
  v8f acc[8] = {vz,vz,vz,vz,vz,vz,vz,vz};
  wave_gemm8<12>(st, LAT_KP, pwlat, acc, lane);

#pragma unroll
  for (int nt = 0; nt < 8; ++nt) {
    const int col = nt * 16 + lo;
    const float bias = blat[col];
#pragma unroll
    for (int r = 0; r < 8; ++r) {
      const int node = n0 + r + hs * 8;
      h[(size_t)node * HDIM + col] = acc[nt][r] + bias;
    }
  }
}

// --------------------------- edge MLP + scatter-mean -----------------------
__global__ __launch_bounds__(128) void edge_kernel(
    const float* __restrict__ h, const float* __restrict__ dis,
    const float* __restrict__ latips, const int* __restrict__ src,
    const int* __restrict__ dst, const int* __restrict__ n2g,
    const unsigned short* __restrict__ pw1, const unsigned short* __restrict__ pw2,
    const float* __restrict__ b1, const float* __restrict__ b2,
    float* __restrict__ agg_sum, float* __restrict__ agg_cnt) {
  __shared__ unsigned short stage[4][16 * EDGE_KP];
  __shared__ unsigned short t1s[4][16 * HDIM];
  const int wave = threadIdx.x >> 5;
  const int lane = threadIdx.x & 31;
  const int lo = lane & 15, hs = lane >> 4;
  const int e0 = (blockIdx.x * 4 + wave) * 16;
  unsigned short* st = stage[wave];

  // Stage einp = [h[src] | h[dst] | dis | lat_ips] as bf16 rows (pad to 352)
  for (int e = 0; e < 16; ++e) {
    const int eid = e0 + e;
    const int s = src[eid];
    const int d = dst[eid];
    float4 hv = *(const float4*)(h + (size_t)s * HDIM + lane * 4);
    v4u16 hp = { f2bf(hv.x), f2bf(hv.y), f2bf(hv.z), f2bf(hv.w) };
    *(v4u16*)(st + e * EDGE_KP + lane * 4) = hp;
    float4 gv = *(const float4*)(h + (size_t)d * HDIM + lane * 4);
    v4u16 gp = { f2bf(gv.x), f2bf(gv.y), f2bf(gv.z), f2bf(gv.w) };
    *(v4u16*)(st + e * EDGE_KP + HDIM + lane * 4) = gp;
    if (lane < 30) {
      float d0 = dis[(size_t)eid * 60 + lane * 2];
      float d1 = dis[(size_t)eid * 60 + lane * 2 + 1];
      v2u16 dp = { f2bf(d0), f2bf(d1) };
      *(v2u16*)(st + e * EDGE_KP + 2 * HDIM + lane * 2) = dp;
    }
    if (lane < 9) {
      const int g = n2g[s];
      st[e * EDGE_KP + 2 * HDIM + 60 + lane] = f2bf(latips[g * 9 + lane]);
    }
    if (lane < EDGE_KP - EDGE_IN) st[e * EDGE_KP + EDGE_IN + lane] = 0;
  }
  __syncthreads();

  const v8f vz = {0.f,0.f,0.f,0.f,0.f,0.f,0.f,0.f};
  v8f acc[8] = {vz,vz,vz,vz,vz,vz,vz,vz};
  wave_gemm8<11>(st, EDGE_KP, pw1, acc, lane);

  unsigned short* t1 = t1s[wave];
#pragma unroll
  for (int nt = 0; nt < 8; ++nt) {
    const int col = nt * 16 + lo;
    const float bias = b1[col];
#pragma unroll
    for (int r = 0; r < 8; ++r)
      t1[(r + hs * 8) * HDIM + col] = f2bf(silu_f(acc[nt][r] + bias));
  }
  __syncthreads();

  v8f acc2[8] = {vz,vz,vz,vz,vz,vz,vz,vz};
  wave_gemm8<4>(t1, HDIM, pw2, acc2, lane);

  int srow[8];
#pragma unroll
  for (int r = 0; r < 8; ++r) srow[r] = src[e0 + r + hs * 8];
#pragma unroll
  for (int nt = 0; nt < 8; ++nt) {
    const int col = nt * 16 + lo;
    const float bias = b2[col];
#pragma unroll
    for (int r = 0; r < 8; ++r) {
      const float v = silu_f(acc2[nt][r] + bias);
      atomicAdd(agg_sum + (size_t)srow[r] * HDIM + col, v);
    }
  }
  if (lane < 16) atomicAdd(agg_cnt + src[e0 + lane], 1.0f);
}

// --------------------------- node MLP + residual + cond --------------------
__global__ __launch_bounds__(128) void node_kernel(
    float* __restrict__ h, const float* __restrict__ agg_sum,
    const float* __restrict__ agg_cnt, const int* __restrict__ n2g,
    const unsigned short* __restrict__ pw1, const unsigned short* __restrict__ pw2,
    const float* __restrict__ b1, const float* __restrict__ b2,
    const float* __restrict__ cbuf) {
  __shared__ unsigned short stage[4][16 * NODE_KP];
  __shared__ unsigned short t1s[4][16 * HDIM];
  const int wave = threadIdx.x >> 5;
  const int lane = threadIdx.x & 31;
  const int lo = lane & 15, hs = lane >> 4;
  const int n0 = (blockIdx.x * 4 + wave) * 16;
  unsigned short* st = stage[wave];

  for (int e = 0; e < 16; ++e) {
    const int node = n0 + e;
    float4 hv = *(const float4*)(h + (size_t)node * HDIM + lane * 4);
    v4u16 hp = { f2bf(hv.x), f2bf(hv.y), f2bf(hv.z), f2bf(hv.w) };
    *(v4u16*)(st + e * NODE_KP + lane * 4) = hp;
    const float inv = 1.0f / fmaxf(agg_cnt[node], 1.0f);
    float4 av = *(const float4*)(agg_sum + (size_t)node * HDIM + lane * 4);
    v4u16 ap = { f2bf(av.x * inv), f2bf(av.y * inv), f2bf(av.z * inv), f2bf(av.w * inv) };
    *(v4u16*)(st + e * NODE_KP + HDIM + lane * 4) = ap;
  }
  __syncthreads();

  const v8f vz = {0.f,0.f,0.f,0.f,0.f,0.f,0.f,0.f};
  v8f acc[8] = {vz,vz,vz,vz,vz,vz,vz,vz};
  wave_gemm8<8>(st, NODE_KP, pw1, acc, lane);

  unsigned short* t1 = t1s[wave];
#pragma unroll
  for (int nt = 0; nt < 8; ++nt) {
    const int col = nt * 16 + lo;
    const float bias = b1[col];
#pragma unroll
    for (int r = 0; r < 8; ++r)
      t1[(r + hs * 8) * HDIM + col] = f2bf(silu_f(acc[nt][r] + bias));
  }
  __syncthreads();

  v8f acc2[8] = {vz,vz,vz,vz,vz,vz,vz,vz};
  wave_gemm8<4>(t1, HDIM, pw2, acc2, lane);

  int gm[8];
#pragma unroll
  for (int r = 0; r < 8; ++r) gm[r] = n2g[n0 + r + hs * 8];
#pragma unroll
  for (int nt = 0; nt < 8; ++nt) {
    const int col = nt * 16 + lo;
    const float bias = b2[col];
#pragma unroll
    for (int r = 0; r < 8; ++r) {
      const int node = n0 + r + hs * 8;
      const float nout = silu_f(acc2[nt][r] + bias);
      const size_t idx = (size_t)node * HDIM + col;
      h[idx] = h[idx] + nout + cbuf[(size_t)gm[r] * HDIM + col];
    }
  }
}

// --------------------------- output heads ----------------------------------
__global__ void gscatter_kernel(const float* __restrict__ h, const int* __restrict__ n2g,
                                float* __restrict__ gf_sum, float* __restrict__ gf_cnt) {
  int idx = blockIdx.x * blockDim.x + threadIdx.x;
  if (idx >= NATOMS * HDIM) return;
  int n = idx >> 7, c = idx & 127;
  int g = n2g[n];
  atomicAdd(gf_sum + (size_t)g * HDIM + c, h[idx]);
  if (c == 0) atomicAdd(gf_cnt + g, 1.0f);
}

__global__ void lattice_kernel(const float* __restrict__ gf_sum, const float* __restrict__ gf_cnt,
                               const float* __restrict__ Wlat, const float* __restrict__ lattices,
                               float* __restrict__ out) {
  int idx = blockIdx.x * blockDim.x + threadIdx.x;
  if (idx >= NGRAPH * 9) return;
  int b = idx / 9, r = idx % 9, i = r / 3, k = r % 3;
  float inv = 1.0f / fmaxf(gf_cnt[b], 1.0f);
  float o = 0.f;
#pragma unroll
  for (int j = 0; j < 3; ++j) {
    float t = 0.f;
    for (int c = 0; c < HDIM; ++c)
      t += gf_sum[(size_t)b * HDIM + c] * Wlat[c * 9 + i * 3 + j];
    o += (t * inv) * lattices[b * 9 + j * 3 + k];
  }
  out[idx] = o;
}

__global__ void coord_kernel(const float* __restrict__ h, const float* __restrict__ Wc,
                             float* __restrict__ out) {
  int idx = blockIdx.x * blockDim.x + threadIdx.x;
  if (idx >= NATOMS * 3) return;
  int n = idx / 3, k = idx % 3;
  float s = 0.f;
  for (int c = 0; c < HDIM; ++c) s += h[(size_t)n * HDIM + c] * Wc[c * 3 + k];
  out[idx] = s;
}

// ---------------------------------------------------------------------------
extern "C" void kernel_launch(void* const* d_in, const int* in_sizes, int n_in,
                              void* d_out, int out_size, void* d_ws, size_t ws_size,
                              hipStream_t stream) {
  const int*   atom_types = (const int*)d_in[0];
  const float* frac       = (const float*)d_in[1];
  const float* lattices   = (const float*)d_in[2];
  const float* z          = (const float*)d_in[3];
  const float* cemb       = (const float*)d_in[4];
  const int*   eidx       = (const int*)d_in[5];
  const int*   src        = eidx;
  const int*   dst        = eidx + NEDGE;
  const int*   n2g        = (const int*)d_in[6];
  const float* emb        = (const float*)d_in[7];
  const float* Wlatent    = (const float*)d_in[8];
  const float* blatent    = (const float*)d_in[9];
  const float* ew1        = (const float*)d_in[10];
  const float* eb1        = (const float*)d_in[11];
  const float* ew2        = (const float*)d_in[12];
  const float* eb2        = (const float*)d_in[13];
  const float* nw1        = (const float*)d_in[14];
  const float* nb1        = (const float*)d_in[15];
  const float* nw2        = (const float*)d_in[16];
  const float* nb2        = (const float*)d_in[17];
  const float* mixin      = (const float*)d_in[18];
  const float* aw1        = (const float*)d_in[19];
  const float* ab1        = (const float*)d_in[20];
  const float* aw2        = (const float*)d_in[21];
  const float* ab2        = (const float*)d_in[22];
  const float* Wc         = (const float*)d_in[23];
  const float* Wlat       = (const float*)d_in[24];

  char* ws = (char*)d_ws;
  size_t off = 0;
  auto alloc = [&](size_t bytes) -> void* {
    void* p = ws + off;
    off = (off + bytes + 255) & ~(size_t)255;
    return p;
  };
  float* dis     = (float*)alloc((size_t)NEDGE * 60 * 4);
  float* latips  = (float*)alloc((size_t)NGRAPH * 9 * 4);
  float* h       = (float*)alloc((size_t)NATOMS * HDIM * 4);
  float* agg_sum = (float*)alloc((size_t)NATOMS * HDIM * 4);
  float* agg_cnt = (float*)alloc((size_t)NATOMS * 4);
  float* cbuf    = (float*)alloc((size_t)NLAYER * NGRAPH * HDIM * 4);
  float* gf_sum  = (float*)alloc((size_t)NGRAPH * HDIM * 4);
  float* gf_cnt  = (float*)alloc((size_t)NGRAPH * 4);
  unsigned short* pw_lat = (unsigned short*)alloc((size_t)12 * 4096 * 2);
  unsigned short* pw_ew1 = (unsigned short*)alloc((size_t)NLAYER * 11 * 4096 * 2);
  unsigned short* pw_ew2 = (unsigned short*)alloc((size_t)NLAYER * 4  * 4096 * 2);
  unsigned short* pw_nw1 = (unsigned short*)alloc((size_t)NLAYER * 8  * 4096 * 2);
  unsigned short* pw_nw2 = (unsigned short*)alloc((size_t)NLAYER * 4  * 4096 * 2);
  (void)in_sizes; (void)n_in; (void)ws_size; (void)out_size;

  // precompute: sinusoid features, lattice inner products, packed weights, cond
  dis_kernel<<<(NEDGE + 255) / 256, 256, 0, stream>>>(frac, src, dst, dis);
  latips_kernel<<<(NGRAPH * 9 + 255) / 256, 256, 0, stream>>>(lattices, latips);

  { dim3 g((12 * 4096 + 255) / 256, 1);      pack_kernel<<<g, 256, 0, stream>>>(Wlatent, pw_lat, LAT_KP, 12); }
  { dim3 g((11 * 4096 + 255) / 256, NLAYER); pack_kernel<<<g, 256, 0, stream>>>(ew1, pw_ew1, EDGE_IN, 11); }
  { dim3 g((4  * 4096 + 255) / 256, NLAYER); pack_kernel<<<g, 256, 0, stream>>>(ew2, pw_ew2, HDIM, 4); }
  { dim3 g((8  * 4096 + 255) / 256, NLAYER); pack_kernel<<<g, 256, 0, stream>>>(nw1, pw_nw1, NODE_KP, 8); }
  { dim3 g((4  * 4096 + 255) / 256, NLAYER); pack_kernel<<<g, 256, 0, stream>>>(nw2, pw_nw2, HDIM, 4); }

  adapt_kernel<<<NLAYER * NGRAPH, HDIM, 0, stream>>>(cemb, aw1, ab1, aw2, ab2, mixin, cbuf);

  h0_kernel<<<NATOMS / 64, 128, 0, stream>>>(atom_types, emb, z, n2g, pw_lat, blatent, h);

  for (int l = 0; l < NLAYER; ++l) {
    hipMemsetAsync(agg_sum, 0, (size_t)NATOMS * HDIM * 4, stream);
    hipMemsetAsync(agg_cnt, 0, (size_t)NATOMS * 4, stream);
    edge_kernel<<<NEDGE / 64, 128, 0, stream>>>(
        h, dis, latips, src, dst, n2g,
        pw_ew1 + (size_t)l * 11 * 4096, pw_ew2 + (size_t)l * 4 * 4096,
        eb1 + l * HDIM, eb2 + l * HDIM, agg_sum, agg_cnt);
    node_kernel<<<NATOMS / 64, 128, 0, stream>>>(
        h, agg_sum, agg_cnt, n2g,
        pw_nw1 + (size_t)l * 8 * 4096, pw_nw2 + (size_t)l * 4 * 4096,
        nb1 + l * HDIM, nb2 + l * HDIM, cbuf + (size_t)l * NGRAPH * HDIM);
  }

  float* out = (float*)d_out;
  hipMemsetAsync(gf_sum, 0, (size_t)NGRAPH * HDIM * 4, stream);
  hipMemsetAsync(gf_cnt, 0, (size_t)NGRAPH * 4, stream);
  gscatter_kernel<<<(NATOMS * HDIM + 255) / 256, 256, 0, stream>>>(h, n2g, gf_sum, gf_cnt);
  lattice_kernel<<<(NGRAPH * 9 + 255) / 256, 256, 0, stream>>>(gf_sum, gf_cnt, Wlat, lattices, out);
  coord_kernel<<<(NATOMS * 3 + 255) / 256, 256, 0, stream>>>(h, Wc, out + NGRAPH * 9);
}
